// DynamicChunker_81595788689816
// MI455X (gfx1250) — compile-verified, matrix operand order checked by reading, not run on hardware
//
#include <hip/hip_runtime.h>
#include <hip/hip_bf16.h>
#include <math.h>

// ---------------------------------------------------------------------------
// DynamicChunker for MI455X (gfx1250, wave32, WMMA f32_16x16x32_f16)
//
// grid = B blocks, 256 threads (8 waves). Each block runs one batch element's
// sequential chunking scan. Per step:
//   - stage x = [cls; window] as f16 in LDS
//   - QKV via WMMA (Q only row 0, K/V rows 0..16)
//   - scores (shared over candidates) + 16 masked softmaxes on VALU
//   - out_proj / FF1 / FF2 / clf1 as M=16 WMMA GEMMs, f32 accum
//   - sigmoid gate, pick first L with p>=THR (or limit-1), emit h2 row
// Weights pre-converted f32->f16 once into d_ws (fits easily in 192MB L2).
// ---------------------------------------------------------------------------

#define Tt   64
#define Dd   256
#define Kk   16
#define Hh   4
#define HD   64
#define FFd  2048
#define EPSc 1e-5f
#define THRc 0.5f
#define SCALEc 0.125f   // 1/sqrt(64)

typedef __attribute__((ext_vector_type(16))) _Float16 v16h;
typedef __attribute__((ext_vector_type(8)))  float    v8f;
typedef __attribute__((ext_vector_type(4)))  int      i32x4;

// f16 fragment: 16 halves per lane as two 16-byte loads (ISA 16-bit A/B layout:
// lane<16 -> k in {k0..k0+7, k0+16..k0+23}; lane>=16 adds +8 via caller).
__device__ inline v16h frag(const _Float16* __restrict__ p) {
  v16h r;
  ((i32x4*)&r)[0] = *(const i32x4*)(p);
  ((i32x4*)&r)[1] = *(const i32x4*)(p + 16);
  return r;
}

// Generic tiled GEMM: C[M x 16*Ntiles] = A(M x 32*Ktiles, f16 LDS) * B^T rows
// (f16 global, row-major [N][Kdim]) + bias + addN, optional relu.
// One wave per output tile, waves stride over Mtiles*Ntiles tiles.
__device__ inline void gemm16(const _Float16* __restrict__ A, int As,
                              int Mtiles, int Mclip,
                              const _Float16* __restrict__ B, int Bs, int Nrow0,
                              int Ntiles, int Ktiles,
                              float* __restrict__ C, _Float16* __restrict__ Ch,
                              int Cs, int Ncol0,
                              const float* __restrict__ bias, int bias0,
                              const float* __restrict__ addN,
                              bool relu, int tid) {
  const int lane = tid & 31;
  const int l15  = lane & 15;
  const int hi8  = (lane & 16) >> 1;      // +8 half-offset for upper half-wave
  const int ntot = Mtiles * Ntiles;
  for (int t = tid >> 5; t < ntot; t += 8) {
    const int mt = t / Ntiles, nt = t % Ntiles;
    v8f acc = {};
    for (int kt = 0; kt < Ktiles; ++kt) {
      v16h a = frag(A + (mt * 16 + l15) * As + kt * 32 + hi8);
      v16h b = frag(B + (Nrow0 + nt * 16 + l15) * Bs + kt * 32 + hi8);
      acc = __builtin_amdgcn_wmma_f32_16x16x32_f16(false, a, false, b,
                                                   (short)0, acc, false, false);
    }
    const int n = nt * 16 + l15;
    float add = (bias ? bias[bias0 + n] : 0.f) + (addN ? addN[n] : 0.f);
#pragma unroll
    for (int r = 0; r < 8; ++r) {
      int m = mt * 16 + hi8 + r;
      if (m < Mclip) {
        float v = acc[r] + add;
        if (relu) v = fmaxf(v, 0.f);
        if (C)  C[m * Cs + Ncol0 + n]  = v;
        if (Ch) Ch[m * Cs + Ncol0 + n] = (_Float16)v;
      }
    }
  }
}

// LayerNorm over 16 rows of 256 (in-place safe). mu/rs are small LDS scratch.
__device__ inline void ln_rows(const float* __restrict__ src,
                               float* __restrict__ dst,
                               _Float16* __restrict__ dsth,
                               const float* __restrict__ w,
                               const float* __restrict__ b,
                               float* mu, float* rs, int tid) {
  if (tid < 16) {
    float s = 0.f, s2 = 0.f;
    for (int c = 0; c < Dd; ++c) { float v = src[tid * Dd + c]; s += v; s2 += v * v; }
    float m = s * (1.f / Dd);
    float var = s2 * (1.f / Dd) - m * m;
    mu[tid] = m; rs[tid] = rsqrtf(var + EPSc);
  }
  __syncthreads();
  for (int idx = tid; idx < 16 * Dd; idx += 256) {
    int r = idx >> 8, c = idx & 255;
    float v = (src[idx] - mu[r]) * rs[r] * w[c] + b[c];
    dst[idx] = v;
    if (dsth) dsth[idx] = (_Float16)v;
  }
  __syncthreads();
}

__global__ void f32_to_f16_kernel(const float* __restrict__ s,
                                  _Float16* __restrict__ d, int n) {
  int i = blockIdx.x * blockDim.x + threadIdx.x;
  if (i < n) d[i] = (_Float16)s[i];
}

__global__ void __launch_bounds__(256)
dynamic_chunker_kernel(const float* __restrict__ frames,
                       const float* __restrict__ clsTok,
                       const _Float16* __restrict__ wIn,   const float* __restrict__ inB,
                       const _Float16* __restrict__ wOut,  const float* __restrict__ outB,
                       const float* __restrict__ ln1w, const float* __restrict__ ln1b,
                       const _Float16* __restrict__ wL1,   const float* __restrict__ l1b,
                       const _Float16* __restrict__ wL2,   const float* __restrict__ l2b,
                       const float* __restrict__ ln2w, const float* __restrict__ ln2b,
                       const _Float16* __restrict__ wC1,   const float* __restrict__ c1b,
                       const float* __restrict__ c2w,  const float* __restrict__ c2b,
                       float* __restrict__ out) {
  __shared__ __attribute__((aligned(16))) _Float16 xf16[32 * Dd];   // 16 KB
  __shared__ __attribute__((aligned(16))) float    kvG[17 * 512];   // 34 KB (K|V, later clf hidden)
  __shared__ __attribute__((aligned(16))) float    q0[Dd];
  __shared__ float    sc[Hh * 17];
  __shared__ __attribute__((aligned(16))) _Float16 Of16[16 * Dd];   // attn out per candidate
  __shared__ __attribute__((aligned(16))) float    buf1[16 * Dd];   // attn+res -> h2
  __shared__ __attribute__((aligned(16))) float    buf2[16 * Dd];   // h1 (normed)
  __shared__ __attribute__((aligned(16))) _Float16 hf16[16 * Dd];   // h1 f16 -> h2 f16
  __shared__ __attribute__((aligned(16))) _Float16 ff1[16 * 512];   // FF1 K-chunk ping buffer
  __shared__ float mu[16], rs[16], pbuf[16];
  __shared__ int   sel;

  const int b = blockIdx.x, tid = threadIdx.x;
  const int lane = tid & 31, wav = tid >> 5;
  const int l15 = lane & 15, hi8 = (lane & 16) >> 1;

  // zero our batch's output region (chunks are zero past the last real chunk)
  for (int i = tid; i < Tt * Dd; i += 256) out[b * Tt * Dd + i] = 0.f;

  int start = 0, cnt = 0;
  while (start < Tt) {
    // ---- stage x = [cls; padded window] as f16 --------------------------
    for (int idx = tid; idx < 17 * Dd; idx += 256) {
      int r = idx >> 8, c = idx & 255;
      float v;
      if (r == 0) v = clsTok[c];
      else { int f = start + r - 1; v = (f < Tt) ? frames[(b * Tt + f) * Dd + c] : 0.f; }
      xf16[idx] = (_Float16)v;
    }
    __syncthreads();

    // ---- QKV: Q row0 only; K/V rows 0..16 -------------------------------
    gemm16(xf16, Dd, 1, 1,  wIn, Dd, 0,   16, 8, q0,  nullptr, Dd,  0,   inB, 0,   nullptr, false, tid);
    gemm16(xf16, Dd, 2, 17, wIn, Dd, 256, 16, 8, kvG, nullptr, 512, 0,   inB, 256, nullptr, false, tid);
    gemm16(xf16, Dd, 2, 17, wIn, Dd, 512, 16, 8, kvG, nullptr, 512, 256, inB, 512, nullptr, false, tid);
    __syncthreads();

    // ---- shared scores: s[h][j] = SCALE * q0_h . k_j_h ------------------
    if (tid < Hh * 17) {
      int h = tid / 17, j = tid % 17;
      const float* qh = q0 + h * HD;
      const float* kh = kvG + j * 512 + h * HD;
      float s = 0.f;
      for (int d = 0; d < HD; ++d) s += qh[d] * kh[d];
      sc[h * 17 + j] = s * SCALEc;
    }
    __syncthreads();

    // ---- 16 masked softmaxes + weighted V sums (VALU) -------------------
    {
      int g  = tid >> 2;          // (L-1)*4 + h, 0..63
      int L  = (g >> 2) + 1;      // 1..16
      int h  = g & 3;
      int d0 = (tid & 3) * 16;
      float m = -1e30f;
      for (int j = 0; j <= L; ++j) m = fmaxf(m, sc[h * 17 + j]);
      float e[17], den = 0.f;
      for (int j = 0; j <= L; ++j) { e[j] = __expf(sc[h * 17 + j] - m); den += e[j]; }
      float inv = 1.f / den;
      for (int dd = 0; dd < 16; ++dd) {
        int d = d0 + dd;
        float o = 0.f;
        for (int j = 0; j <= L; ++j) o += e[j] * kvG[j * 512 + 256 + h * HD + d];
        Of16[(L - 1) * Dd + h * HD + d] = (_Float16)(o * inv);
      }
    }
    __syncthreads();

    // ---- out_proj + residual(cls) -> LN1 --------------------------------
    gemm16(Of16, Dd, 1, 16, wOut, Dd, 0, 16, 8, buf1, nullptr, Dd, 0, outB, 0, clsTok, false, tid);
    __syncthreads();
    ln_rows(buf1, buf2, hf16, ln1w, ln1b, mu, rs, tid);      // h1 -> buf2/hf16

    // ---- FFN: FF1 chunked (K=512), FF2 accumulated in VGPRs -------------
    v8f acc0 = {}, acc1 = {};
    for (int c = 0; c < 4; ++c) {
      gemm16(hf16, Dd, 1, 16, wL1, Dd, c * 512, 32, 8,
             nullptr, ff1, 512, 0, l1b, c * 512, nullptr, true, tid);
      __syncthreads();
      for (int kt = 0; kt < 16; ++kt) {
        v16h a  = frag(ff1 + l15 * 512 + kt * 32 + hi8);
        int  kc = c * 512 + kt * 32 + hi8;
        v16h b0 = frag(wL2 + (wav * 16 + l15) * FFd + kc);
        acc0 = __builtin_amdgcn_wmma_f32_16x16x32_f16(false, a, false, b0, (short)0, acc0, false, false);
        v16h b1 = frag(wL2 + ((wav + 8) * 16 + l15) * FFd + kc);
        acc1 = __builtin_amdgcn_wmma_f32_16x16x32_f16(false, a, false, b1, (short)0, acc1, false, false);
      }
      __syncthreads();
    }
#pragma unroll
    for (int t = 0; t < 2; ++t) {
      int n = (wav + t * 8) * 16 + l15;
      v8f acc = t ? acc1 : acc0;
#pragma unroll
      for (int r = 0; r < 8; ++r) {
        int m = hi8 + r;
        buf1[m * Dd + n] = acc[r] + l2b[n] + buf2[m * Dd + n];  // ff + h1 residual
      }
    }
    __syncthreads();
    ln_rows(buf1, buf1, hf16, ln2w, ln2b, mu, rs, tid);      // h2 f32 in buf1, f16 in hf16

    // ---- classifier: relu(h2 @ clf1^T + b) . clf2 + b -> sigmoid --------
    gemm16(hf16, Dd, 1, 16, wC1, Dd, 0, 16, 8, kvG, nullptr, Dd, 0, c1b, 0, nullptr, true, tid);
    __syncthreads();
    if (tid < 16) {
      float a = c2b[0];
      for (int c = 0; c < Dd; ++c) a += kvG[tid * Dd + c] * c2w[c];
      pbuf[tid] = 1.f / (1.f + __expf(-a));
    }
    __syncthreads();
    if (tid == 0) {
      int limit = (Tt - start < Kk) ? (Tt - start) : Kk;
      int idx = limit - 1;
      for (int L = 1; L <= Kk; ++L)
        if (pbuf[L - 1] >= THRc && L <= limit) { idx = L - 1; break; }
      sel = idx;
    }
    __syncthreads();
    int idx = sel;
    for (int c = tid; c < Dd; c += 256)
      out[(b * Tt + cnt) * Dd + c] = buf1[idx * Dd + c];
    start += idx + 1;
    cnt   += 1;
    __syncthreads();
  }
}

// f16 weight arena offsets inside d_ws (halves)
#define OFF_INW  0u
#define OFF_OUTW (OFF_INW  + 768u * 256u)
#define OFF_L1   (OFF_OUTW + 256u * 256u)
#define OFF_L2   (OFF_L1   + 2048u * 256u)
#define OFF_C1   (OFF_L2   + 256u * 2048u)
#define WS_HALVES (OFF_C1  + 256u * 256u)

extern "C" void kernel_launch(void* const* d_in, const int* in_sizes, int n_in,
                              void* d_out, int out_size, void* d_ws, size_t ws_size,
                              hipStream_t stream) {
  if (ws_size < (size_t)WS_HALVES * sizeof(_Float16)) return;  // scratch too small

  const float* frames = (const float*)d_in[0];
  const float* clsTok = (const float*)d_in[1];
  const float* inW    = (const float*)d_in[2];
  const float* inB    = (const float*)d_in[3];
  const float* outW   = (const float*)d_in[4];
  const float* outB   = (const float*)d_in[5];
  const float* ln1w   = (const float*)d_in[6];
  const float* ln1b   = (const float*)d_in[7];
  const float* l1W    = (const float*)d_in[8];
  const float* l1b    = (const float*)d_in[9];
  const float* l2W    = (const float*)d_in[10];
  const float* l2b    = (const float*)d_in[11];
  const float* ln2w   = (const float*)d_in[12];
  const float* ln2b   = (const float*)d_in[13];
  const float* c1W    = (const float*)d_in[14];
  const float* c1b    = (const float*)d_in[15];
  const float* c2w    = (const float*)d_in[16];
  const float* c2b    = (const float*)d_in[17];

  _Float16* ws = (_Float16*)d_ws;
  struct { const float* s; _Float16* d; int n; } cv[5] = {
    { inW,  ws + OFF_INW,  768 * 256 },
    { outW, ws + OFF_OUTW, 256 * 256 },
    { l1W,  ws + OFF_L1,  2048 * 256 },
    { l2W,  ws + OFF_L2,  256 * 2048 },
    { c1W,  ws + OFF_C1,   256 * 256 },
  };
  for (int i = 0; i < 5; ++i)
    f32_to_f16_kernel<<<(cv[i].n + 255) / 256, 256, 0, stream>>>(cv[i].s, cv[i].d, cv[i].n);

  int B = in_sizes[0] / (Tt * Dd);
  dynamic_chunker_kernel<<<B, 256, 0, stream>>>(
      frames, clsTok,
      ws + OFF_INW, inB, ws + OFF_OUTW, outB,
      ln1w, ln1b,
      ws + OFF_L1, l1b, ws + OFF_L2, l2b,
      ln2w, ln2b,
      ws + OFF_C1, c1b, c2w, c2b,
      (float*)d_out);
}
